// customLSTM_74406013436392
// MI455X (gfx1250) — matrix operand order repaired and checked by reading, not compiled, not run on hardware
//
#include <hip/hip_runtime.h>
#include <hip/hip_bf16.h>

// ---------------------------------------------------------------------------
// Types & helpers
// ---------------------------------------------------------------------------
typedef unsigned short bf16_t;                                   // raw bf16 storage
typedef __attribute__((ext_vector_type(16))) __bf16 v16bf;       // WMMA A/B frag
typedef __attribute__((ext_vector_type(8)))  float  v8f;         // WMMA C/D frag
typedef __attribute__((ext_vector_type(4)))  unsigned int u32x4; // TDM D# group0
typedef __attribute__((ext_vector_type(8)))  int  i32x8;         // TDM D# group1
typedef __attribute__((ext_vector_type(4)))  int  i32x4;         // TDM D# group2/3

#if defined(__HIP_DEVICE_COMPILE__) && defined(__gfx1250__)
#if __has_builtin(__builtin_amdgcn_tensor_load_to_lds) && \
    __has_builtin(__builtin_amdgcn_s_wait_tensorcnt)
#define USE_TDM 1
#endif
#endif

__device__ __forceinline__ bf16_t f2bf(float f) {
  union { float f; unsigned u; } v; v.f = f;
  unsigned r = v.u + 0x7FFFu + ((v.u >> 16) & 1u);               // round-nearest-even
  return (bf16_t)(r >> 16);
}

#define TN 64
#define TK 32
#define LDSPAD 8                 // +8 bf16 -> 80-byte row stride (16B aligned)

// ---------------------------------------------------------------------------
// Generic bf16 WMMA GEMM:  C[M,N] = act( A[M,K](bf16) @ B[K,N](bf16)
//                                        + bias0 + bias1 + Cinit )
// Block = 128 threads = 4 waves; block tile (64*RPW) x 64;
// wave tile (16*RPW) x 64 -> RPW*4 back-to-back WMMAs per K step.
// ALL loops touching the accumulators are fully unrolled so acc[][] is
// scalarized into VGPRs (no scratch spills around the WMMA burst).
// ---------------------------------------------------------------------------
template <int RPW>
__global__ __launch_bounds__(128)
void gemm_bf16(const bf16_t* __restrict__ A, int lda,
               const bf16_t* __restrict__ Bw, int ldb,
               const float* __restrict__ bias0, const float* __restrict__ bias1,
               const float* __restrict__ Cinit, int ldci,
               float* __restrict__ Cf, bf16_t* __restrict__ Cb, int ldc,
               int M, int N, int K, int act)
{
  constexpr int TME = 64 * RPW;            // rows per block
  __shared__ bf16_t As[TME][TK + LDSPAD];  // row-major (rows of A)
  __shared__ bf16_t Bs[TN][TK + LDSPAD];   // TRANSPOSED (rows = columns of B)

  const int tid    = threadIdx.x;
  const int wave   = tid >> 5;
  const int lane   = tid & 31;
  const int lane16 = lane & 15;
  const int hi     = lane >> 4;            // 0: lanes 0-15, 1: lanes 16-31
  const int m0     = blockIdx.y * TME;
  const int n0     = blockIdx.x * TN;
  const int waveRow0 = m0 + wave * (16 * RPW);   // first row of this wave's strip

  union Frag { v16bf v; bf16_t u[16]; uint4 q[2]; };
  union Acc  { v8f  v; float  f[8];  };

  Acc acc[RPW][4];
  #pragma unroll
  for (int r = 0; r < RPW; ++r) {
    #pragma unroll
    for (int ct = 0; ct < 4; ++ct) {
      int col = n0 + ct * 16 + lane16;
      float bv = 0.f;
      if (col < N) {
        if (bias0) bv += bias0[col];
        if (bias1) bv += bias1[col];
      }
      #pragma unroll
      for (int e = 0; e < 8; ++e) {
        int row = waveRow0 + r * 16 + e + hi * 8;   // CDNA5 32-bit C/D layout
        float v = bv;
        if (Cinit && row < M && col < N) v += Cinit[(size_t)row * ldci + col];
        acc[r][ct].f[e] = v;
      }
    }
  }

#ifdef USE_TDM
  const unsigned ldsAs = (unsigned)(uintptr_t)&As[0][0];
#endif

  for (int k0 = 0; k0 < K; k0 += TK) {
    const bool fullK = (k0 + TK <= K);
    const bool fullA = fullK && (m0 + TME <= M);
    const bool fullB = fullK && (n0 + TN <= N) && ((ldb & 7) == 0);

    // ---- stage A tile (TME x 32, row-major, padded rows) ----
#ifdef USE_TDM
    if (fullA) {
      // One descriptor per wave: 16*RPW x 32 sub-tile. TDM inserts the
      // 16-byte row pad (pad_interval = 16 DWORDs -> 3, pad_amount = 4 -> 3).
      const size_t gaddr =
          (size_t)(uintptr_t)&A[(size_t)(m0 + wave * 16 * RPW) * lda + k0];
      u32x4 g0;
      g0[0] = 1u;                                            // count=1
      g0[1] = ldsAs + (unsigned)(wave * 16 * RPW * (TK + LDSPAD) * 2);
      g0[2] = (unsigned)(gaddr & 0xFFFFFFFFu);
      g0[3] = (unsigned)((gaddr >> 32) & 0x1FFFFFFu) | (2u << 30);  // type=2
      i32x8 g1;
      g1[0] = (1 << 16) | (1 << 20) | (3 << 22) | (3 << 25); // 2B, pad en/intvl/amt
      g1[1] = (TK & 0xFFFF) << 16;                           // tensor_dim0 = 32
      g1[2] = ((16 * RPW) & 0xFFFF) << 16;                   // tensor_dim1
      g1[3] = (TK & 0xFFFF) << 16;                           // tile_dim0   = 32
      g1[4] = 16 * RPW;                                      // tile_dim1
      g1[5] = lda;                                           // dim0 stride (elems)
      g1[6] = 0;
      g1[7] = 0;
      i32x4 gz = {0, 0, 0, 0};
#if __clang_major__ >= 23
      i32x8 gz8 = {0, 0, 0, 0, 0, 0, 0, 0};
      __builtin_amdgcn_tensor_load_to_lds(g0, g1, gz, gz, gz8, 0);
#else
      __builtin_amdgcn_tensor_load_to_lds(g0, g1, gz, gz, 0);
#endif
    } else
#endif
    if (fullA && ((lda & 7) == 0)) {
      for (int i = tid; i < (TME * TK) / 8; i += 128) {
        int r  = i >> 2;
        int c8 = (i & 3) * 8;
        *(uint4*)&As[r][c8] =
            *(const uint4*)&A[(size_t)(m0 + r) * lda + (k0 + c8)];
      }
    } else {
      for (int i = tid; i < TME * TK; i += 128) {
        int r = i >> 5, cc = i & (TK - 1);
        int gr = m0 + r, gc = k0 + cc;
        As[r][cc] = (gr < M && gc < K) ? A[(size_t)gr * lda + gc] : (bf16_t)0;
      }
    }

    // ---- stage B tile (32 x 64) transposed into Bs[n][k] ----
    if (fullB) {
      for (int i = tid; i < (TK * TN) / 8; i += 128) {
        int r  = i >> 3;                   // k row 0..31
        int c8 = (i & 7) * 8;              // n column group
        const uint4 q = *(const uint4*)&Bw[(size_t)(k0 + r) * ldb + (n0 + c8)];
        const bf16_t* e = (const bf16_t*)&q;
        #pragma unroll
        for (int j = 0; j < 8; ++j) Bs[c8 + j][r] = e[j];
      }
    } else {
      for (int i = tid; i < TK * TN; i += 128) {
        int r = i >> 6, cc = i & (TN - 1);
        int gr = k0 + r, gc = n0 + cc;
        Bs[cc][r] = (gr < K && gc < N) ? Bw[(size_t)gr * ldb + gc] : (bf16_t)0;
      }
    }

    // speculative prefetch of next K tile (global_prefetch_b8)
    if (k0 + TK < K) {
      __builtin_prefetch(&A[(size_t)(m0 + (tid >> 1)) * lda + (k0 + TK)], 0, 1);
      __builtin_prefetch(&Bw[(size_t)(k0 + TK + (tid >> 2)) * ldb + n0], 0, 1);
    }

#ifdef USE_TDM
    if (fullA) __builtin_amdgcn_s_wait_tensorcnt(0);
#endif
    __syncthreads();

    // ---- load ALL fragments first (ds_load_b128 x (2*RPW + 8)) ----
    Frag af[RPW];
    #pragma unroll
    for (int r = 0; r < RPW; ++r) {
      int ar = wave * 16 * RPW + r * 16 + lane16;
      int kb = hi * 8;
      af[r].q[0] = *(const uint4*)&As[ar][kb];
      af[r].q[1] = *(const uint4*)&As[ar][16 + kb];
    }
    Frag bf4[4];
    #pragma unroll
    for (int ct = 0; ct < 4; ++ct) {
      int bn = ct * 16 + lane16;
      int kb = hi * 8;
      bf4[ct].q[0] = *(const uint4*)&Bs[bn][kb];
      bf4[ct].q[1] = *(const uint4*)&Bs[bn][16 + kb];
    }

    // ---- WMMA burst: RPW*4 back-to-back matrix ops ----
    #pragma unroll
    for (int ct = 0; ct < 4; ++ct) {
      #pragma unroll
      for (int r = 0; r < RPW; ++r) {
        acc[r][ct].v = __builtin_amdgcn_wmma_f32_16x16x32_bf16(
            /*neg_a=*/false, af[r].v, /*neg_b=*/false, bf4[ct].v,
            /*c_mod=*/(short)0, acc[r][ct].v, /*reuse_a=*/false, /*reuse_b=*/false);
      }
    }
    __syncthreads();
  }

  #pragma unroll
  for (int r = 0; r < RPW; ++r) {
    #pragma unroll
    for (int ct = 0; ct < 4; ++ct) {
      int col = n0 + ct * 16 + lane16;
      if (col < N) {
        #pragma unroll
        for (int e = 0; e < 8; ++e) {
          int row = waveRow0 + r * 16 + e + hi * 8;
          if (row < M) {
            float v = acc[r][ct].f[e];
            if (act == 1) v = v > 0.f ? v : 0.f;
            size_t idx = (size_t)row * ldc + col;
            if (Cf) Cf[idx] = v;
            if (Cb) Cb[idx] = f2bf(v);
          }
        }
      }
    }
  }
}

// ---------------------------------------------------------------------------
// Small utility kernels
// ---------------------------------------------------------------------------
// fp32 (rows x cols, dense) -> bf16 (rows x ldd, padded)
__global__ void f32_to_bf16_2d_k(const float* __restrict__ s, bf16_t* __restrict__ d,
                                 int rows, int cols, int ldd) {
  int i = blockIdx.x * blockDim.x + threadIdx.x;
  if (i >= rows * cols) return;
  int r = i / cols, c = i - r * cols;
  d[(size_t)r * ldd + c] = f2bf(s[i]);
}

__global__ void zero_state_k(float* __restrict__ h, float* __restrict__ c,
                             bf16_t* __restrict__ hb, int nf, int nb) {
  int i = blockIdx.x * blockDim.x + threadIdx.x;
  if (i < nf) { h[i] = 0.f; c[i] = 0.f; }
  if (i < nb) hb[i] = 0;
}

// gates (i,f,o,g) -> (h_t, c_t); writes fp32 h/c for recurrence, bf16 h (padded
// leading dim) for the next step's WMMA A operand and for the head GEMMs.
__global__ void lstm_point_k(const float* __restrict__ G,
                             float* __restrict__ h, float* __restrict__ c,
                             bf16_t* __restrict__ hb, bf16_t* __restrict__ lo,
                             int Bn, int Hn, int ldh)
{
  int i = blockIdx.x * blockDim.x + threadIdx.x;
  if (i >= Bn * Hn) return;
  int b = i / Hn, j = i - b * Hn;
  const float* g = G + (size_t)b * 4 * Hn;
  float it = 1.f / (1.f + expf(-g[j]));
  float ft = 1.f / (1.f + expf(-g[Hn + j]));
  float ot = 1.f / (1.f + expf(-g[2 * Hn + j]));
  float gt = tanhf(g[3 * Hn + j]);
  float ct = ft * c[i] + it * gt;
  float ht = ot * tanhf(ct);
  c[i] = ct; h[i] = ht;
  bf16_t hv = f2bf(ht);
  size_t pidx = (size_t)b * ldh + j;
  hb[pidx] = hv; lo[pidx] = hv;
}

// quaternion normalize + rotate 4 pattern points + translate; writes the
// three outputs (xq, xp, rotated) concatenated flat into d_out.
__global__ void quat_rot_k(const float* __restrict__ qraw, const float* __restrict__ praw,
                           const float* __restrict__ patterns, float* __restrict__ out,
                           int Mtot)
{
  int i = blockIdx.x * blockDim.x + threadIdx.x;
  if (i >= Mtot) return;
  float q0 = qraw[i * 4 + 0], q1 = qraw[i * 4 + 1];
  float q2 = qraw[i * 4 + 2], q3 = qraw[i * 4 + 3];
  float inv = rsqrtf(q0 * q0 + q1 * q1 + q2 * q2 + q3 * q3);
  q0 *= inv; q1 *= inv; q2 *= inv; q3 *= inv;
  float px = praw[i * 3 + 0], py = praw[i * 3 + 1], pz = praw[i * 3 + 2];

  out[(size_t)i * 4 + 0] = q0;
  out[(size_t)i * 4 + 1] = q1;
  out[(size_t)i * 4 + 2] = q2;
  out[(size_t)i * 4 + 3] = q3;
  float* xp = out + (size_t)4 * Mtot + (size_t)i * 3;
  xp[0] = px; xp[1] = py; xp[2] = pz;

  float* rout = out + (size_t)7 * Mtot + (size_t)i * 12;
  #pragma unroll
  for (int k = 0; k < 4; ++k) {
    const float* v = patterns + (size_t)i * 12 + k * 3;
    float vx = v[0], vy = v[1], vz = v[2];
    float ux = q2 * vz - q3 * vy;          // uv  = qvec x v
    float uy = q3 * vx - q1 * vz;
    float uz = q1 * vy - q2 * vx;
    float wx = q2 * uz - q3 * uy;          // uuv = qvec x uv
    float wy = q3 * ux - q1 * uz;
    float wz = q1 * uy - q2 * ux;
    rout[k * 3 + 0] = vx + 2.f * (q0 * ux + wx) + px;
    rout[k * 3 + 1] = vy + 2.f * (q0 * uy + wy) + py;
    rout[k * 3 + 2] = vz + 2.f * (q0 * uz + wz) + pz;
  }
}

// ---------------------------------------------------------------------------
// Launch
// ---------------------------------------------------------------------------
extern "C" void kernel_launch(void* const* d_in, const int* in_sizes, int n_in,
                              void* d_out, int out_size, void* d_ws, size_t ws_size,
                              hipStream_t stream)
{
  (void)in_sizes; (void)n_in; (void)out_size; (void)ws_size;
  const int S = 199, B = 1024, H = 150;
  const size_t Mtot = (size_t)S * B;
  const int Mt = (int)Mtot;
  const int LDH = 152;                     // padded leading dim for H=150

  // ---- inputs (setup_inputs dict order; params tuples flattened w,b) ----
  const float* det      = (const float*)d_in[0];
  const float* patterns = (const float*)d_in[1];
  const float* fc1w = (const float*)d_in[2],  *fc1b = (const float*)d_in[3];
  const float* fc2w = (const float*)d_in[4],  *fc2b = (const float*)d_in[5];
  const float* fc3w = (const float*)d_in[6],  *fc3b = (const float*)d_in[7];
  const float* fc4w = (const float*)d_in[8],  *fc4b = (const float*)d_in[9];
  const float* fc5w = (const float*)d_in[10], *fc5b = (const float*)d_in[11];
  const float* i2hw = (const float*)d_in[12], *i2hb = (const float*)d_in[13];
  const float* h2hw = (const float*)d_in[14], *h2hb = (const float*)d_in[15];
  const float* q1w  = (const float*)d_in[16], *q1b  = (const float*)d_in[17];
  const float* q2w  = (const float*)d_in[18], *q2b  = (const float*)d_in[19];
  const float* q3w  = (const float*)d_in[20], *q3b  = (const float*)d_in[21];
  const float* p1w  = (const float*)d_in[22], *p1b  = (const float*)d_in[23];
  const float* p2w  = (const float*)d_in[24], *p2b  = (const float*)d_in[25];
  const float* p3w  = (const float*)d_in[26], *p3b  = (const float*)d_in[27];

  // ---- workspace carve-out ----
  size_t off = 0;
  auto alloc = [&](size_t bytes) -> void* {
    size_t a = (off + 255) & ~(size_t)255;
    off = a + bytes;
    return (void*)((char*)d_ws + a);
  };
  // bf16 weights, leading dims padded to multiples of 8 elements
  bf16_t* wb_fc1 = (bf16_t*)alloc((size_t)12  * 304 * 2);
  bf16_t* wb_fc2 = (bf16_t*)alloc((size_t)300 * 400 * 2);
  bf16_t* wb_fc3 = (bf16_t*)alloc((size_t)400 * 504 * 2);
  bf16_t* wb_fc4 = (bf16_t*)alloc((size_t)500 * 352 * 2);
  bf16_t* wb_fc5 = (bf16_t*)alloc((size_t)350 * 304 * 2);
  bf16_t* wb_i2h = (bf16_t*)alloc((size_t)300 * 600 * 2);
  bf16_t* wb_h2h = (bf16_t*)alloc((size_t)150 * 600 * 2);
  bf16_t* wb_q1  = (bf16_t*)alloc((size_t)150 * 104 * 2);
  bf16_t* wb_q2  = (bf16_t*)alloc((size_t)100 * 56  * 2);
  bf16_t* wb_q3  = (bf16_t*)alloc((size_t)50  * 8   * 2);
  bf16_t* wb_p1  = (bf16_t*)alloc((size_t)150 * 152 * 2);
  bf16_t* wb_p2  = (bf16_t*)alloc((size_t)150 * 56  * 2);
  bf16_t* wb_p3  = (bf16_t*)alloc((size_t)50  * 8   * 2);

  bf16_t* det_bf = (bf16_t*)alloc(Mtot * 16  * 2);   // ld 16
  bf16_t* ping   = (bf16_t*)alloc(Mtot * 512 * 2);   // activations up to ld 504
  bf16_t* pong   = (bf16_t*)alloc(Mtot * 512 * 2);
  float*  P      = (float*) alloc(Mtot * 600 * 4);   // hoisted X@Wi + (bi+bh)
  float*  G      = (float*) alloc((size_t)B * 600 * 4);
  float*  h      = (float*) alloc((size_t)B * H * 4);
  float*  c      = (float*) alloc((size_t)B * H * 4);
  bf16_t* h_bf   = (bf16_t*)alloc((size_t)B * LDH * 2);
  bf16_t* lstm_o = (bf16_t*)alloc(Mtot * LDH * 2);
  bf16_t* xq1    = (bf16_t*)alloc(Mtot * 104 * 2);
  bf16_t* xq2    = (bf16_t*)alloc(Mtot * 56  * 2);
  float*  qraw   = (float*) alloc(Mtot * 4   * 4);
  bf16_t* xp1    = (bf16_t*)alloc(Mtot * 152 * 2);
  bf16_t* xp2    = (bf16_t*)alloc(Mtot * 56  * 2);
  float*  praw   = (float*) alloc(Mtot * 3   * 4);

  auto conv2 = [&](const float* s, bf16_t* d, size_t rows, int cols, int ldd) {
    size_t n = rows * (size_t)cols;
    f32_to_bf16_2d_k<<<(unsigned)((n + 255) / 256), 256, 0, stream>>>(
        s, d, (int)rows, cols, ldd);
  };
  // rpw=2 for the big S*B-row GEMMs (fewer blocks, better LDS/WMMA ratio);
  // rpw=1 for the per-step LSTM GEMM (more blocks -> occupancy across WGPs).
  auto gemm = [&](const bf16_t* A, int lda, const bf16_t* Bw, int ldb,
                  const float* b0, const float* b1,
                  const float* Ci, int ldci,
                  float* Cf, bf16_t* Cb, int ldc,
                  int M, int N, int K, int act, int rpw) {
    if (rpw == 2) {
      dim3 grid((N + TN - 1) / TN, (M + 127) / 128);
      gemm_bf16<2><<<grid, 128, 0, stream>>>(A, lda, Bw, ldb, b0, b1, Ci, ldci,
                                             Cf, Cb, ldc, M, N, K, act);
    } else {
      dim3 grid((N + TN - 1) / TN, (M + 63) / 64);
      gemm_bf16<1><<<grid, 128, 0, stream>>>(A, lda, Bw, ldb, b0, b1, Ci, ldci,
                                             Cf, Cb, ldc, M, N, K, act);
    }
  };

  // ---- weight / input conversion to bf16 (padded leading dims) ----
  conv2(fc1w, wb_fc1, 12,  300, 304);  conv2(fc2w, wb_fc2, 300, 400, 400);
  conv2(fc3w, wb_fc3, 400, 500, 504);  conv2(fc4w, wb_fc4, 500, 350, 352);
  conv2(fc5w, wb_fc5, 350, 300, 304);  conv2(i2hw, wb_i2h, 300, 600, 600);
  conv2(h2hw, wb_h2h, 150, 600, 600);  conv2(q1w,  wb_q1,  150, 100, 104);
  conv2(q2w,  wb_q2,  100, 50,  56);   conv2(q3w,  wb_q3,  50,  4,   8);
  conv2(p1w,  wb_p1,  150, 150, 152);  conv2(p2w,  wb_p2,  150, 50,  56);
  conv2(p3w,  wb_p3,  50,  3,   8);
  conv2(det,  det_bf, Mtot, 12, 16);
  {
    int nf = B * H, nb = B * LDH;
    zero_state_k<<<(nb + 255) / 256, 256, 0, stream>>>(h, c, h_bf, nf, nb);
  }

  // ---- MLP stack (batch GEMMs, relu, bf16 activations) ----
  gemm(det_bf, 16,  wb_fc1, 304, fc1b, nullptr, nullptr, 0, nullptr, ping, 304, Mt, 300, 12,  1, 2);
  gemm(ping,   304, wb_fc2, 400, fc2b, nullptr, nullptr, 0, nullptr, pong, 400, Mt, 400, 300, 1, 2);
  gemm(pong,   400, wb_fc3, 504, fc3b, nullptr, nullptr, 0, nullptr, ping, 504, Mt, 500, 400, 1, 2);
  gemm(ping,   504, wb_fc4, 352, fc4b, nullptr, nullptr, 0, nullptr, pong, 352, Mt, 350, 500, 1, 2);
  gemm(pong,   352, wb_fc5, 304, fc5b, nullptr, nullptr, 0, nullptr, ping, 304, Mt, 300, 350, 1, 2);

  // ---- hoisted input projection for all timesteps: P = X@Wi + (bi + bh) ----
  gemm(ping, 304, wb_i2h, 600, i2hb, h2hb, nullptr, 0, P, nullptr, 600, Mt, 600, 300, 0, 2);

  // ---- LSTM recurrence: only h@Wh stays sequential ----
  for (int t = 0; t < S; ++t) {
    gemm(h_bf, LDH, wb_h2h, 600, nullptr, nullptr,
         P + (size_t)t * B * 600, 600, G, nullptr, 600, B, 600, H, 0, 1);
    lstm_point_k<<<(B * H + 255) / 256, 256, 0, stream>>>(
        G, h, c, h_bf, lstm_o + (size_t)t * B * LDH, B, H, LDH);
  }

  // ---- heads ----
  gemm(lstm_o, LDH, wb_q1, 104, q1b, nullptr, nullptr, 0, nullptr, xq1, 104, Mt, 100, H,   1, 2);
  gemm(xq1,    104, wb_q2, 56,  q2b, nullptr, nullptr, 0, nullptr, xq2, 56,  Mt, 50,  100, 1, 2);
  gemm(xq2,    56,  wb_q3, 8,   q3b, nullptr, nullptr, 0, qraw, nullptr, 4,  Mt, 4,   50,  0, 2);
  gemm(lstm_o, LDH, wb_p1, 152, p1b, nullptr, nullptr, 0, nullptr, xp1, 152, Mt, 150, H,   1, 2);
  gemm(xp1,    152, wb_p2, 56,  p2b, nullptr, nullptr, 0, nullptr, xp2, 56,  Mt, 50,  150, 1, 2);
  gemm(xp2,    56,  wb_p3, 8,   p3b, nullptr, nullptr, 0, praw, nullptr, 3,  Mt, 3,   50,  0, 2);

  // ---- quaternion normalize + rotate + pack outputs ----
  quat_rot_k<<<(unsigned)((Mtot + 255) / 256), 256, 0, stream>>>(
      qraw, praw, patterns, (float*)d_out, Mt);
}
